// DecoderAttention_65335042506788
// MI455X (gfx1250) — matrix operand order repaired and checked by reading, not compiled, not run on hardware
//
#include <hip/hip_runtime.h>
#include <hip/hip_bf16.h>

#define NTOK   768
#define DMODEL 384
#define DPAIR  128
#define NH     8
#define HS     32
#define HSALL  256   // NH*HS
#define CCAT   512   // 2*NH*HS
#define EPS    1e-5f

typedef __attribute__((ext_vector_type(16))) __bf16 v16bf;
typedef __attribute__((ext_vector_type(8)))  __bf16 bf16x8;
typedef __attribute__((ext_vector_type(8)))  float  v8f;

union AFrag { v16bf v; bf16x8 h[2]; };

__device__ inline float wave_sum8(float x){
  #pragma unroll
  for(int m=1;m<8;m<<=1) x += __shfl_xor(x,m,32);
  return x;
}
__device__ inline float wave_sum32(float x){
  #pragma unroll
  for(int m=1;m<32;m<<=1) x += __shfl_xor(x,m,32);
  return x;
}
__device__ inline float wave_max32(float x){
  #pragma unroll
  for(int m=1;m<32;m<<=1) x = fmaxf(x,__shfl_xor(x,m,32));
  return x;
}

// ---------------- LayerNorm of local [768,384] ----------------
__global__ void k_ln_local(const float* __restrict__ x, float* __restrict__ y){
  int i = blockIdx.x, tid = threadIdx.x;          // 128 threads
  float v0 = x[i*DMODEL + tid];
  float v1 = x[i*DMODEL + tid + 128];
  float v2 = x[i*DMODEL + tid + 256];
  float s = v0+v1+v2, s2 = v0*v0+v1*v1+v2*v2;
  s = wave_sum32(s); s2 = wave_sum32(s2);
  __shared__ float red[2][4];
  int w = tid>>5, l = tid&31;
  if(l==0){ red[0][w]=s; red[1][w]=s2; }
  __syncthreads();
  s  = red[0][0]+red[0][1]+red[0][2]+red[0][3];
  s2 = red[1][0]+red[1][1]+red[1][2]+red[1][3];
  float mu   = s*(1.0f/DMODEL);
  float var  = s2*(1.0f/DMODEL) - mu*mu;
  float rstd = rsqrtf(var + EPS);
  y[i*DMODEL+tid]     = (v0-mu)*rstd;
  y[i*DMODEL+tid+128] = (v1-mu)*rstd;
  y[i*DMODEL+tid+256] = (v2-mu)*rstd;
}

// ---------------- QKV projection + per-head LN (wave32 == head) ----------------
__global__ void k_qkv(const float* __restrict__ ln,
                      const float* __restrict__ Wq, const float* __restrict__ bq,
                      const float* __restrict__ Wk, const float* __restrict__ bk,
                      const float* __restrict__ Wv, const float* __restrict__ bv,
                      float* __restrict__ q, float* __restrict__ k, float* __restrict__ v){
  int i = blockIdx.x, t = threadIdx.x;            // 256 threads
  __shared__ float sl[DMODEL];
  sl[t] = ln[i*DMODEL + t];
  if(t < DMODEL-256) sl[t+256] = ln[i*DMODEL + t + 256];
  __syncthreads();
  float aq = bq[t], ak = bk[t], av = bv[t];
  for(int a=0;a<DMODEL;a++){
    float x = sl[a];
    aq = fmaf(x, Wq[a*HSALL+t], aq);
    ak = fmaf(x, Wk[a*HSALL+t], ak);
    av = fmaf(x, Wv[a*HSALL+t], av);
  }
  // head h == wave (32 consecutive threads): LN over S=32 via shuffles
  float sq = wave_sum32(aq), sq2 = wave_sum32(aq*aq);
  float muq = sq*(1.0f/HS), vq = sq2*(1.0f/HS) - muq*muq;
  float qn = (aq-muq)*rsqrtf(vq+EPS) * rsqrtf((float)HS + 1e-6f);
  float sk = wave_sum32(ak), sk2 = wave_sum32(ak*ak);
  float muk = sk*(1.0f/HS), vk = sk2*(1.0f/HS) - muk*muk;
  float kn = (ak-muk)*rsqrtf(vk+EPS);
  q[i*HSALL+t] = qn; k[i*HSALL+t] = kn; v[i*HSALL+t] = av;
}

// ---------------- Pack [Wpb|Wpv] (128x40 -> 128x48 padded) into WMMA B-fragment order ----------------
// layout: [kt(4)][nt(3)][lane(32)][e(16)], B elem: col = nt*16+(lane&15), K = kt*32+(lane>>4)*8+(e&7)+(e>>3)*16
__global__ void k_pack_pairw(const float* __restrict__ Wpb, const float* __restrict__ Wpv,
                             __bf16* __restrict__ out){
  int idx = blockIdx.x*blockDim.x + threadIdx.x;
  if(idx >= 4*3*512) return;
  int e = idx & 15, lane = (idx>>4)&31;
  int nt = (idx>>9)%3, kt = (idx>>9)/3;
  int col = nt*16 + (lane&15);
  int K   = kt*32 + (lane>>4)*8 + (e&7) + (e>>3)*16;
  float val = 0.0f;
  if(col < NH)          val = Wpb[K*NH + col];
  else if(col < NH+HS)  val = Wpv[K*HS + (col-NH)];
  out[idx] = (__bf16)val;
}

// ---------------- Pack Wo [512,384] into WMMA B-fragment order [kt(16)][nt(24)][lane][e] ----------------
__global__ void k_pack_wo(const float* __restrict__ Wo, __bf16* __restrict__ out){
  int idx = blockIdx.x*blockDim.x + threadIdx.x;
  if(idx >= 16*24*512) return;
  int e = idx & 15, lane = (idx>>4)&31;
  int nt = (idx>>9)%24, kt = (idx>>9)/24;
  int col = nt*16 + (lane&15);
  int K   = kt*32 + (lane>>4)*8 + (e&7) + (e>>3)*16;
  out[idx] = (__bf16)Wo[K*DMODEL + col];
}

// ---------------- Fused pair LN + dual projection (the 302MB-read kernel) ----------------
// 16 flattened (i,j) rows per block; 128 threads = 4 waves; waves 0..2 each own one
// 16-col N-tile of the 16x48 output (cols 0..7 = bias, 8..39 = pair-value, 40..47 pad).
__global__ void k_pair(const float* __restrict__ pair, const __bf16* __restrict__ Bw,
                       float* __restrict__ biasT,   // [i][h][j]
                       float* __restrict__ pv){     // [i*N+j][S]
  int r0  = blockIdx.x * 16;                 // flattened ij base (768%16==0: tile stays in one i)
  int tid = threadIdx.x;
  __shared__ __align__(16) float  sp[16][DPAIR];
  __shared__ __align__(16) __bf16 sa[16][DPAIR];
  // stage 16x128 f32 tile
  const float* gsrc = pair + (size_t)r0 * DPAIR;
  float* spf = &sp[0][0];
  #pragma unroll
  for(int p=0;p<4;p++){
    int fi = p*512 + tid*4;
    *(float4*)(spf + fi) = *(const float4*)(gsrc + fi);
  }
  __syncthreads();
  // LN over 128, 8 lanes per row (xor-shuffle reduce inside 8-lane group)
  int row = tid>>3, sub = tid&7;
  float s=0.f, s2=0.f;
  #pragma unroll
  for(int e=0;e<16;e++){ float x = sp[row][sub*16+e]; s += x; s2 += x*x; }
  s = wave_sum8(s); s2 = wave_sum8(s2);
  float mu   = s*(1.0f/DPAIR);
  float var  = s2*(1.0f/DPAIR) - mu*mu;
  float rstd = rsqrtf(var + EPS);
  #pragma unroll
  for(int e=0;e<16;e++){
    int c = sub*16+e;
    sa[row][c] = (__bf16)((sp[row][c]-mu)*rstd);
  }
  __syncthreads();
  int w = tid>>5, lane = tid&31;
  if(w < 3){                                  // wave-uniform branch: EXEC all-ones for WMMA
    int arow = lane & 15, koff = (lane>>4)*8; // bf16 A 16x32 fragment layout
    v8f acc = {0,0,0,0,0,0,0,0};
    #pragma unroll
    for(int kt=0;kt<4;kt++){                  // K = 128 = 4 x 32
      AFrag A, B;
      A.h[0] = *(const bf16x8*)&sa[arow][kt*32 + koff];
      A.h[1] = *(const bf16x8*)&sa[arow][kt*32 + koff + 16];
      const __bf16* bp = Bw + ((size_t)(kt*3 + w)*32 + lane)*16;
      B.h[0] = *(const bf16x8*)(bp);
      B.h[1] = *(const bf16x8*)(bp + 8);
      acc = __builtin_amdgcn_wmma_f32_16x16x32_bf16(false, A.v, false, B.v,
                                                    (short)0, acc, false, false);
    }
    int gi = r0 / NTOK, j0 = r0 - gi*NTOK;
    int mbase = (lane>>4)*8;
    int col   = w*16 + (lane&15);
    #pragma unroll
    for(int r=0;r<8;r++){
      float val = acc[r];
      int jr = j0 + mbase + r;
      if(col < NH)
        biasT[((size_t)gi*NH + col)*NTOK + jr] = val;
      else if(col < NH+HS)
        pv[((size_t)(r0 + mbase + r))*HS + (col-NH)] = val;
    }
  }
}

// ---------------- Online-softmax attention: block per query row, wave32 per head ----------------
__global__ void k_attn(const float* __restrict__ q, const float* __restrict__ k,
                       const float* __restrict__ v, const float* __restrict__ biasT,
                       const float* __restrict__ pv, const unsigned char* __restrict__ mask,
                       __bf16* __restrict__ cc){
  int i = blockIdx.x;
  int tid = threadIdx.x, h = tid>>5, lane = tid&31;
  __shared__ float sqh[NH][HS];
  sqh[h][lane] = q[i*HSALL + tid];
  __syncthreads();
  bool maski = (mask[i] != 0);
  float m = -1e30f, ssum = 0.0f, acc1 = 0.0f, acc2 = 0.0f;
  for(int c=0;c<NTOK/32;c++){
    int j = c*32 + lane;
    float logit = biasT[((size_t)i*NH + h)*NTOK + j];
    const float* kr = k + ((size_t)j*NH + h)*HS;
    float d = 0.f;
    #pragma unroll
    for(int a=0;a<HS;a++) d = fmaf(sqh[h][a], kr[a], d);
    logit += d;
    bool valid = maski && (mask[j] != 0);
    logit = valid ? logit : -1e9f;
    float cm = wave_max32(logit);
    float nm = fmaxf(m, cm);
    float wexp  = __expf(logit - nm);
    float scale = __expf(m - nm);
    ssum = ssum*scale + wave_sum32(wexp);
    acc1 *= scale; acc2 *= scale;
    float wm = valid ? wexp : 0.0f;           // post-softmax mask folded in
    #pragma unroll
    for(int l=0;l<32;l++){
      float wl = __shfl(wm, l, 32);
      int jj = c*32 + l;
      acc1 = fmaf(wl, v [((size_t)jj*NH + h)*HS + lane], acc1);
      acc2 = fmaf(wl, pv[((size_t)i*NTOK + jj)*HS + lane], acc2);
    }
    m = nm;
  }
  float inv = 1.0f / ssum;
  cc[(size_t)i*CCAT + h*HS + lane]         = (__bf16)(acc1*inv);
  cc[(size_t)i*CCAT + HSALL + h*HS + lane] = (__bf16)(acc2*inv);
}

// ---------------- Output projection [768,512] @ [512,384] via WMMA, 1 wave / 16x16 tile ----------------
__global__ void k_out(const __bf16* __restrict__ cc, const __bf16* __restrict__ Bw,
                      float* __restrict__ out){
  int mt = blockIdx.x, nt = blockIdx.y;
  int lane = threadIdx.x;
  int arow = lane&15, koff = (lane>>4)*8;
  const __bf16* arowp = cc + (size_t)(mt*16 + arow)*CCAT;
  v8f acc = {0,0,0,0,0,0,0,0};
  #pragma unroll
  for(int kt=0;kt<16;kt++){                   // K = 512 = 16 x 32
    AFrag A, B;
    A.h[0] = *(const bf16x8*)(arowp + kt*32 + koff);
    A.h[1] = *(const bf16x8*)(arowp + kt*32 + koff + 16);
    const __bf16* bp = Bw + ((size_t)(kt*24 + nt)*32 + lane)*16;
    B.h[0] = *(const bf16x8*)(bp);
    B.h[1] = *(const bf16x8*)(bp + 8);
    acc = __builtin_amdgcn_wmma_f32_16x16x32_bf16(false, A.v, false, B.v,
                                                  (short)0, acc, false, false);
  }
  int mbase = mt*16 + (lane>>4)*8;
  int col   = nt*16 + (lane&15);
  #pragma unroll
  for(int r=0;r<8;r++) out[(size_t)(mbase+r)*DMODEL + col] = acc[r];
}

extern "C" void kernel_launch(void* const* d_in, const int* in_sizes, int n_in,
                              void* d_out, int out_size, void* d_ws, size_t ws_size,
                              hipStream_t stream){
  const float* local = (const float*)d_in[0];
  const float* pair  = (const float*)d_in[1];
  const unsigned char* mask = (const unsigned char*)d_in[2];
  const float* Wq  = (const float*)d_in[3];
  const float* bq  = (const float*)d_in[4];
  const float* Wk  = (const float*)d_in[5];
  const float* bk  = (const float*)d_in[6];
  const float* Wv  = (const float*)d_in[7];
  const float* bv  = (const float*)d_in[8];
  const float* Wpb = (const float*)d_in[9];
  const float* Wpv = (const float*)d_in[10];
  const float* Wo  = (const float*)d_in[11];
  float* out = (float*)d_out;

  char* ws = (char*)d_ws;
  size_t off = 0;
  auto alloc = [&](size_t bytes){ void* p = ws + off; off += (bytes + 255) & ~(size_t)255; return p; };
  float*  localn = (float*) alloc((size_t)NTOK*DMODEL*4);
  float*  qb     = (float*) alloc((size_t)NTOK*HSALL*4);
  float*  kb     = (float*) alloc((size_t)NTOK*HSALL*4);
  float*  vb     = (float*) alloc((size_t)NTOK*HSALL*4);
  float*  biasT  = (float*) alloc((size_t)NTOK*NH*NTOK*4);
  float*  pvb    = (float*) alloc((size_t)NTOK*NTOK*HS*4);
  __bf16* cc     = (__bf16*)alloc((size_t)NTOK*CCAT*2);
  __bf16* pwPair = (__bf16*)alloc((size_t)4*3*512*2);
  __bf16* pwO    = (__bf16*)alloc((size_t)16*24*512*2);

  hipLaunchKernelGGL(k_ln_local,   dim3(NTOK),            dim3(128), 0, stream, local, localn);
  hipLaunchKernelGGL(k_pack_pairw, dim3((4*3*512+255)/256),  dim3(256), 0, stream, Wpb, Wpv, pwPair);
  hipLaunchKernelGGL(k_pack_wo,    dim3((16*24*512+255)/256),dim3(256), 0, stream, Wo, pwO);
  hipLaunchKernelGGL(k_qkv,        dim3(NTOK),            dim3(256), 0, stream,
                     localn, Wq,bq, Wk,bk, Wv,bv, qb,kb,vb);
  hipLaunchKernelGGL(k_pair,       dim3(NTOK*NTOK/16),    dim3(128), 0, stream, pair, pwPair, biasT, pvb);
  hipLaunchKernelGGL(k_attn,       dim3(NTOK),            dim3(256), 0, stream,
                     qb,kb,vb, biasT, pvb, mask, cc);
  hipLaunchKernelGGL(k_out,        dim3(48,24),           dim3(32),  0, stream, cc, pwO, out);
}
